// TurboQuantAttention_25108378812704
// MI455X (gfx1250) — compile-verified, matrix operand order checked by reading, not run on hardware
//
#include <hip/hip_runtime.h>
#include <math.h>

#define BATCH   2
#define NHEADS  32
#define SEQ     2048
#define HDIM    128
#define QMAXF   127.0f

#define QTILE   128      // query rows per workgroup (8 waves x 16 rows)
#define KBLK    64       // key/value tokens per LDS block
#define KPAD    136      // halves: 128 dims + 8 pad  (272B rows; TDM pad 64dw+4dw)
#define VPAD    72       // halves: 64 tokens + 8 pad (144B rows; TDM pad 32dw+4dw)
#define PPAD    72       // halves: P tile stride

typedef __attribute__((ext_vector_type(16))) _Float16 v16h;
typedef __attribute__((ext_vector_type(8)))  _Float16 v8h;
typedef __attribute__((ext_vector_type(4)))  _Float16 v4h;
typedef __attribute__((ext_vector_type(8)))  float    v8f;
typedef __attribute__((ext_vector_type(4)))  float    v4f;
typedef __attribute__((ext_vector_type(4)))  unsigned int u32x4;
typedef __attribute__((ext_vector_type(8)))  int      i32x8;
typedef __attribute__((ext_vector_type(4)))  int      i32x4;

union V16H { v16h v; v8h h[2]; };

// ---------- reductions (wave32; halves 0-15 / 16-31 stay separate for masks<=8) ----
__device__ __forceinline__ float hmax16(float v) {
    v = fmaxf(v, __shfl_xor(v, 8, 32));
    v = fmaxf(v, __shfl_xor(v, 4, 32));
    v = fmaxf(v, __shfl_xor(v, 2, 32));
    v = fmaxf(v, __shfl_xor(v, 1, 32));
    return v;
}
__device__ __forceinline__ float hsum16(float v) {
    v += __shfl_xor(v, 8, 32);
    v += __shfl_xor(v, 4, 32);
    v += __shfl_xor(v, 2, 32);
    v += __shfl_xor(v, 1, 32);
    return v;
}
__device__ __forceinline__ float wmax32(float v) {
    v = fmaxf(v, __shfl_xor(v, 16, 32));
    return hmax16(v);
}

// ---------- 16-bit A/B fragment load per ISA 7.12.2 ------------------------------
// lanes 0-15:  row = lane,    K = {0..7, 16..23}  (two aligned 16B LDS loads)
// lanes 16-31: row = lane-16, K = {8..15, 24..31}
__device__ __forceinline__ v16h ldfrag(const _Float16* base, int strideH,
                                       int r16, int hi) {
    const _Float16* p = base + r16 * strideH + hi * 8;
    V16H u;
    u.h[0] = *(const v8h*)(p);
    u.h[1] = *(const v8h*)(p + 16);
    return u.v;
}

// ---------- TDM: 2-D tile DMA global -> LDS with hardware row padding -------------
// Descriptor packed per CDNA5 ISA 8.3/8.4: group0 {count=1, lds_addr, global_addr,
// type=2}; group1 {data_size=2B, pad ctl, tensor/tile dims, dim0 stride}; remaining
// descriptor groups zero (2-D tile: tile_dim2/3 = 0).  6-arg clang-23 builtin form.
__device__ __forceinline__ void tdm_load_2d(const void* gptr, void* lptr,
                                            unsigned tensor_d0, unsigned tensor_d1,
                                            unsigned stride0_elems,
                                            unsigned tile_d0, unsigned tile_d1,
                                            unsigned pad_interval_code,
                                            unsigned pad_amount_code) {
    unsigned long long ga  = (unsigned long long)gptr;
    unsigned           lds = (unsigned)(unsigned long long)lptr; // low 32b = LDS offset
    u32x4 g0;
    g0[0] = 1u;                                        // count=1, user descriptor
    g0[1] = lds;                                       // lds_addr [63:32]
    g0[2] = (unsigned)(ga & 0xffffffffu);              // global_addr [95:64]
    g0[3] = (unsigned)((ga >> 32) & 0x01ffffffu)       // global_addr [120:96]
          | (2u << 30);                                // type=2 (image) [127:126]
    i32x8 g1;
    g1[0] = (int)((1u << 16)                           // data_size = 2 bytes
                | (1u << 20)                           // pad_enable
                | (pad_interval_code << 22)
                | (pad_amount_code  << 25));
    g1[1] = (int)((tensor_d0 & 0xffffu) << 16);        // tensor_dim0[15:0]
    g1[2] = (int)(((tensor_d0 >> 16) & 0xffffu) | ((tensor_d1 & 0xffffu) << 16));
    g1[3] = (int)(((tensor_d1 >> 16) & 0xffffu) | ((tile_d0 & 0xffffu) << 16));
    g1[4] = (int)(tile_d1 & 0xffffu);                  // tile_dim1; tile_dim2=0
    g1[5] = (int)stride0_elems;                        // tensor_dim0_stride[31:0]
    g1[6] = 0;
    g1[7] = 0;
    i32x4 z4 = {0, 0, 0, 0};
    i32x8 z8 = {0, 0, 0, 0, 0, 0, 0, 0};
    __builtin_amdgcn_tensor_load_to_lds(g0, g1, z4, z4, z8, 0);
}

// ================= Phase 1: fake-quant K/V -> dequantized f16 in ws ===============
// one wave per (b,h,token) row of 128 dims; lane handles 4 floats.
// K written row-major [bh][token][dim]; V written TRANSPOSED [bh][dim][token]
// (scatter stores merge in the 192MB L2 since Vt fits entirely).
__global__ __launch_bounds__(256)
void tq_quant_kv(const float* __restrict__ K, const float* __restrict__ V,
                 _Float16* __restrict__ Kq, _Float16* __restrict__ Vt) {
    const int wave = threadIdx.x >> 5;
    const int lane = threadIdx.x & 31;
    const long row = (long)blockIdx.x * 8 + wave;       // global (bh, token) row
    const int  isV = blockIdx.y;

    const float* p = (isV ? V : K) + row * HDIM + lane * 4;
    v4f x = *(const v4f*)p;

    float amax = fmaxf(fmaxf(fabsf(x[0]), fabsf(x[1])),
                       fmaxf(fabsf(x[2]), fabsf(x[3])));
    amax = wmax32(amax);

    float scale = fmaxf(amax * (1.0f / QMAXF), 1e-8f);
    float inv   = 1.0f / scale;

    _Float16 q[4];
#pragma unroll
    for (int i = 0; i < 4; ++i) {
        float qi = rintf(x[i] * inv);                  // round-half-even == jnp.round
        qi = fminf(fmaxf(qi, -QMAXF), QMAXF);
        q[i] = (_Float16)(qi * scale);                 // dequant; int*scale exact in f16
    }

    if (!isV) {
        v4h qq = { q[0], q[1], q[2], q[3] };
        *(v4h*)(Kq + row * HDIM + lane * 4) = qq;
    } else {
        const long bh  = row / SEQ;
        const long tok = row - bh * SEQ;
        _Float16* dst = Vt + bh * (long)HDIM * SEQ + tok;
#pragma unroll
        for (int i = 0; i < 4; ++i)
            dst[(long)(lane * 4 + i) * SEQ] = q[i];
    }
}

// ================= Phase 2: flash attention over f16 K / transposed V =============
__global__ __launch_bounds__(256)
void tq_attn(const float* __restrict__ Q, const _Float16* __restrict__ Kq,
             const _Float16* __restrict__ Vt, float* __restrict__ Out) {
    __shared__ __align__(16) _Float16 sK [KBLK * KPAD];      // [token][dim]  via TDM
    __shared__ __align__(16) _Float16 sVt[HDIM * VPAD];      // [dim][token]  via TDM
    __shared__ __align__(16) _Float16 sP [8 * 16 * PPAD];    // per-wave P tiles

    const int tid  = threadIdx.x;
    const int wave = tid >> 5;
    const int lane = tid & 31;
    const int r16  = lane & 15;
    const int hi   = lane >> 4;

    const int  bh    = blockIdx.x >> 4;                      // SEQ/QTILE == 16
    const int  qt    = blockIdx.x & 15;
    const long base  = (long)bh * SEQ * HDIM;
    const int  qrow0 = qt * QTILE + wave * 16;

    // ---- Q fragments (A operand), f32 -> f16 on the fly, held in VGPRs ----
    v16h qf[4];
    {
        const float* qp = Q + base + (long)(qrow0 + r16) * HDIM + hi * 8;
#pragma unroll
        for (int c = 0; c < 4; ++c) {
            const float* p0 = qp + c * 32;
            v4f a0 = *(const v4f*)(p0);
            v4f a1 = *(const v4f*)(p0 + 4);
            v4f b0 = *(const v4f*)(p0 + 16);
            v4f b1 = *(const v4f*)(p0 + 20);
            V16H u;
#pragma unroll
            for (int i = 0; i < 4; ++i) {
                u.h[0][i]     = (_Float16)a0[i];
                u.h[0][4 + i] = (_Float16)a1[i];
                u.h[1][i]     = (_Float16)b0[i];
                u.h[1][4 + i] = (_Float16)b1[i];
            }
            qf[c] = u.v;
        }
    }

    v8f   o[8];
    float m[8], l[8];
#pragma unroll
    for (int d = 0; d < 8; ++d) o[d] = (v8f)(0.0f);
#pragma unroll
    for (int r = 0; r < 8; ++r) { m[r] = -1.0e30f; l[r] = 0.0f; }

    const float sc = 0.08838834764831845f;   // 1/sqrt(128)
    _Float16* Pw = sP + wave * 16 * PPAD;

#pragma unroll 1
    for (int kb = 0; kb < SEQ / KBLK; ++kb) {
        // ---- TDM: DMA this block's K and Vt tiles into LDS (wave 0 only).
        //      K : 64 tokens x 128 dims, 256B rows + 16B pad  -> KPAD stride
        //      Vt: 128 dims  x 64 toks,  128B rows + 16B pad  -> VPAD stride
        if (wave == 0) {
            tdm_load_2d(Kq + base + (long)kb * KBLK * HDIM, sK,
                        HDIM, SEQ, HDIM, HDIM, KBLK,
                        /*pad: every 64 dw*/5, /*+4 dw*/3);
            tdm_load_2d(Vt + base + (long)kb * KBLK, sVt,
                        SEQ, HDIM, SEQ, KBLK, HDIM,
                        /*pad: every 32 dw*/4, /*+4 dw*/3);
            __builtin_amdgcn_s_wait_tensorcnt(0);
        }
        __syncthreads();

        // ---- S = (Q K^T) * sc : 4 N-tiles x 4 K-chunks of WMMA ----
        v8f s[4];
#pragma unroll
        for (int t = 0; t < 4; ++t) {
            v8f acc = (v8f)(0.0f);
#pragma unroll
            for (int c = 0; c < 4; ++c) {
                v16h bf = ldfrag(sK + (t * 16) * KPAD + c * 32, KPAD, r16, hi);
                acc = __builtin_amdgcn_wmma_f32_16x16x32_f16(
                          false, qf[c], false, bf, (short)0, acc, false, false);
            }
#pragma unroll
            for (int i = 0; i < 8; ++i) acc[i] *= sc;
            s[t] = acc;
        }

        // ---- online softmax (row r+8*hi lives in VGPR index r across 16 lanes) ----
        float alpha[8];
#pragma unroll
        for (int r = 0; r < 8; ++r) {
            float cm = fmaxf(fmaxf(s[0][r], s[1][r]), fmaxf(s[2][r], s[3][r]));
            cm = hmax16(cm);
            float mn = fmaxf(m[r], cm);
            alpha[r] = __expf(m[r] - mn);
            m[r] = mn;
        }
#pragma unroll
        for (int d = 0; d < 8; ++d)
#pragma unroll
            for (int r = 0; r < 8; ++r) o[d][r] *= alpha[r];

        float rs[8];
#pragma unroll
        for (int r = 0; r < 8; ++r) rs[r] = 0.0f;
#pragma unroll
        for (int t = 0; t < 4; ++t)
#pragma unroll
            for (int r = 0; r < 8; ++r) {
                float p = __expf(s[t][r] - m[r]);
                rs[r] += p;
                Pw[(r + 8 * hi) * PPAD + t * 16 + r16] = (_Float16)p;
            }
#pragma unroll
        for (int r = 0; r < 8; ++r)
            l[r] = l[r] * alpha[r] + hsum16(rs[r]);

        // ---- O += P V : 2 K-chunks x 8 dim-tiles of WMMA ----
#pragma unroll
        for (int c2 = 0; c2 < 2; ++c2) {
            v16h pf = ldfrag(Pw + c2 * 32, PPAD, r16, hi);
#pragma unroll
            for (int d = 0; d < 8; ++d) {
                v16h vf = ldfrag(sVt + (d * 16) * VPAD + c2 * 32, VPAD, r16, hi);
                o[d] = __builtin_amdgcn_wmma_f32_16x16x32_f16(
                           false, pf, false, vf, (short)0, o[d], false, false);
            }
        }
        __syncthreads();   // protect sK/sVt before next block's TDM
    }

    // ---- normalize and store ----
#pragma unroll
    for (int r = 0; r < 8; ++r) {
        float invl = 1.0f / l[r];
        long rowoff = base + (long)(qrow0 + r + 8 * hi) * HDIM + r16;
#pragma unroll
        for (int d = 0; d < 8; ++d)
            Out[rowoff + d * 16] = o[d][r] * invl;
    }
}

// ================================ launcher ========================================
extern "C" void kernel_launch(void* const* d_in, const int* in_sizes, int n_in,
                              void* d_out, int out_size, void* d_ws, size_t ws_size,
                              hipStream_t stream) {
    const float* Q = (const float*)d_in[0];
    const float* K = (const float*)d_in[1];
    const float* V = (const float*)d_in[2];
    float*       O = (float*)d_out;

    const long N = (long)BATCH * NHEADS * SEQ * HDIM;   // 16,777,216 elems
    _Float16* Kq = (_Float16*)d_ws;                     // 32 MB, [bh][tok][dim]
    _Float16* Vt = Kq + N;                              // 32 MB, [bh][dim][tok]

    // Phase 1: fake-quant K and V into f16 workspace (one wave per token row)
    dim3 qgrid((BATCH * NHEADS * SEQ) / 8, 2);
    tq_quant_kv<<<qgrid, 256, 0, stream>>>(K, V, Kq, Vt);

    // Phase 2: flash attention, one workgroup per (b,h,q-tile)
    tq_attn<<<BATCH * NHEADS * (SEQ / QTILE), 256, 0, stream>>>(Q, Kq, Vt, O);
}
